// TopoBrainPhysical_5892695130715
// MI455X (gfx1250) — matrix operand order repaired and checked by reading, not compiled, not run on hardware
//
#include <hip/hip_runtime.h>
#include <hip/hip_bf16.h>
#include <math.h>

typedef __attribute__((ext_vector_type(16))) _Float16 v16h;
typedef __attribute__((ext_vector_type(8)))  _Float16 v8h;
typedef __attribute__((ext_vector_type(4)))  _Float16 v4h;
typedef __attribute__((ext_vector_type(8)))  float    v8f;

#define WPB    8      // waves per block (wave32)
#define BLOCK  256
#define PI_F   3.14159265358979323846f
#define LOG2E_F 1.44269504088896340736f

union V16U { v16h v; v8h h[2]; };

__device__ __forceinline__ void lds_fence() {
  asm volatile("s_wait_dscnt 0" ::: "memory");
}

// branch-free tanh: 1 - 2/(exp2(2*log2e*x)+1); saturates to +/-1 at inf/0
__device__ __forceinline__ float fast_tanh(float x) {
  float e = __builtin_amdgcn_exp2f(x * (2.0f * LOG2E_F));
  return 1.0f - 2.0f * __builtin_amdgcn_rcpf(e + 1.0f);
}
__device__ __forceinline__ float fast_sigmoid(float x) {
  float e = __builtin_amdgcn_exp2f(-x * LOG2E_F);
  return __builtin_amdgcn_rcpf(1.0f + e);
}

// A operand (16x32 f16, MxK): lane holds row M=lane%16.
// halves 0..7  = K [ 8*hi,  8*hi+8) ; halves 8..15 = K [16+8*hi, +8)
__device__ __forceinline__ v16h gatherA(const _Float16* tile, int strideHalves) {
  const int lane = threadIdx.x & 31;
  const int row = lane & 15, hi = lane >> 4;
  const _Float16* p = tile + row * strideHalves + 8 * hi;
  V16U u;
  u.h[0] = *(const v8h*)p;
  u.h[1] = *(const v8h*)(p + 16);
  return u.v;
}

// B operand (32x16 f16, KxN): lane holds column N=lane%16,
// halves 0..7 = K [8*hi, 8*hi+8) ; halves 8..15 = K [16+8*hi, +8)
__device__ __forceinline__ v16h gatherB(const _Float16* tile) {
  const int lane = threadIdx.x & 31;
  const int n = lane & 15, hi = lane >> 4;
  const _Float16* p = tile + n * 32 + hi * 8;
  V16U u;
  u.h[0] = *(const v8h*)p;
  u.h[1] = *(const v8h*)(p + 16);
  return u.v;
}

__device__ __forceinline__ v8f wmma16(v16h a, v16h b, v8f c) {
  return __builtin_amdgcn_wmma_f32_16x16x32_f16(false, a, false, b, (short)0, c,
                                                false, false);
}

__device__ __forceinline__ v8f splat8(float bv) {
  v8f c;
#pragma unroll
  for (int i = 0; i < 8; ++i) c[i] = bv;
  return c;
}

// store two N-tiles of a D accumulator (cols 0..31) into an A-layout staging
// tile [16 rows][32 halves], applying fast tanh.
__device__ __forceinline__ void storePairTanh(_Float16* tile, v8f a0, v8f a1) {
  const int lane = threadIdx.x & 31;
  const int col = lane & 15, hi = lane >> 4;
#pragma unroll
  for (int v = 0; v < 8; ++v) {
    const int row = v + 8 * hi;
    tile[row * 32 + col]      = (_Float16)fast_tanh(a0[v]);
    tile[row * 32 + 16 + col] = (_Float16)fast_tanh(a1[v]);
  }
}

__global__ __launch_bounds__(BLOCK) void topo_fwd(
    const float* __restrict__ x,
    const float* __restrict__ enc_w1, const float* __restrict__ enc_b1,
    const float* __restrict__ enc_w2, const float* __restrict__ enc_b2,
    const float* __restrict__ enc_w3, const float* __restrict__ enc_b3,
    const float* __restrict__ node_w1, const float* __restrict__ node_b1,
    const float* __restrict__ node_w2, const float* __restrict__ node_b2,
    const float* __restrict__ ang_logit, const float* __restrict__ rad_logit,
    const float* __restrict__ read_w1, const float* __restrict__ read_b1,
    const float* __restrict__ read_w2, const float* __restrict__ read_b2,
    float* __restrict__ out, int Bn) {
  __shared__ __align__(16) _Float16 wt[15 * 512];
  __shared__ float biasS[7 * 32];
  __shared__ float MtotS[64];
  __shared__ __align__(16) _Float16 tA[WPB][512];     // zero-padded node input
  __shared__ __align__(16) _Float16 tB[WPB][512];     // transpose staging
  __shared__ __align__(16) _Float16 h3[WPB][16 * 96]; // flat 96-vec per batch
  __shared__ float zb[WPB][64];

  const int tid = threadIdx.x;

  // ---------- stage all weights as f16 B-tiles ----------
  for (int i = tid; i < 15 * 512; i += BLOCK) {
    const int t = i >> 9, rem = i & 511;
    const int n = rem >> 5, k = rem & 31;
    float v = 0.f;
    if (t < 2)        { int col = t * 16 + n;       if (k < 4  && col < 24) v = enc_w1[k * 24 + col]; }
    else if (t < 4)   { int col = (t - 2) * 16 + n; if (k < 24 && col < 24) v = enc_w2[k * 24 + col]; }
    else if (t == 4)  {                             if (k < 24 && n < 3)    v = enc_w3[k * 3 + n]; }
    else if (t < 7)   { int col = (t - 5) * 16 + n; if (k < 12 && col < 24) v = node_w1[k * 24 + col]; }
    else if (t == 7)  {                             if (k < 24 && n < 12)   v = node_w2[k * 12 + n]; }
    else if (t < 14)  { int tt = t - 8; int kk = (tt >> 1) * 32 + k; int col = (tt & 1) * 16 + n;
                        if (kk < 96 && col < 24) v = read_w1[kk * 24 + col]; }
    else              {                             if (k < 24 && n < 4)    v = read_w2[k * 4 + n]; }
    wt[i] = (_Float16)v;
  }
  for (int i = tid; i < 7 * 32; i += BLOCK) {
    const int b = i >> 5, j = i & 31;
    float v = 0.f;
    switch (b) {
      case 0: if (j < 24) v = enc_b1[j]; break;
      case 1: if (j < 24) v = enc_b2[j]; break;
      case 2: if (j < 3)  v = enc_b3[j]; break;
      case 3: if (j < 24) v = node_b1[j]; break;
      case 4: if (j < 12) v = node_b2[j]; break;
      case 5: if (j < 24) v = read_b1[j]; break;
      default: if (j < 4) v = read_b2[j]; break;
    }
    biasS[i] = v;
  }
  if (tid == 0) {
    // normalized ring adjacencies -> fused 8x8 node-mixing matrix
    float sa[4], mx = -1e30f, s = 0.f;
    for (int i = 0; i < 4; ++i) mx = fmaxf(mx, ang_logit[i]);
    for (int i = 0; i < 4; ++i) { sa[i] = expf(ang_logit[i] - mx); s += sa[i]; }
    for (int i = 0; i < 4; ++i) sa[i] /= s;
    float ang[4][4];
    for (int i = 0; i < 4; ++i) {
      float rs = 0.f;
      for (int j = 0; j < 4; ++j) {
        float adj = ((j == ((i + 1) & 3)) || (j == ((i + 3) & 3))) ? 1.f : 0.f;
        ang[i][j] = sa[j] * adj; rs += ang[i][j];
      }
      float den = fmaxf(rs, 1e-6f);
      for (int j = 0; j < 4; ++j) ang[i][j] /= den;
    }
    float sr[2]; mx = fmaxf(rad_logit[0], rad_logit[1]);
    sr[0] = expf(rad_logit[0] - mx); sr[1] = expf(rad_logit[1] - mx);
    s = sr[0] + sr[1]; sr[0] /= s; sr[1] /= s;
    float rad[2][2];
    for (int i = 0; i < 2; ++i) {
      float rs = 0.f;
      for (int j = 0; j < 2; ++j) {
        float adj = (j == (1 - i)) ? 1.f : 0.f;
        rad[i][j] = sr[j] * adj; rs += rad[i][j];
      }
      float den = fmaxf(rs, 1e-6f);
      for (int j = 0; j < 2; ++j) rad[i][j] /= den;
    }
    for (int n = 0; n < 8; ++n)
      for (int m = 0; m < 8; ++m) {
        float v = (n == m) ? 1.f : 0.f;
        if ((m & 1) == (n & 1)) v += ang[n >> 1][m >> 1];
        if ((m & 3) == (n & 3)) v += rad[n >> 2][m >> 2];
        MtotS[n * 8 + m] = v;
      }
  }
  __syncthreads();

  const int lane = tid & 31;
  const int wave = tid >> 5;
  const int col = lane & 15;
  const int hi = lane >> 4;
  _Float16* mytA = tA[wave];
  _Float16* mytB = tB[wave];
  _Float16* myh3 = h3[wave];
  float* myzb = zb[wave];

  // zero tA once: node-net writes only cols 0..3; cols 4..31 stay zero
  {
    v8h z8;
#pragma unroll
    for (int i = 0; i < 8; ++i) z8[i] = (_Float16)0.f;
    *(v8h*)(mytA + lane * 16) = z8;
    *(v8h*)(mytA + lane * 16 + 8) = z8;
  }

  // hoist loop-invariant operands into registers
  const v16h wN1a = gatherB(wt + 5 * 512);
  const v16h wN1b = gatherB(wt + 6 * 512);
  const v16h wN2  = gatherB(wt + 7 * 512);
  const float bE1a = biasS[0 * 32 + col], bE1b = biasS[0 * 32 + 16 + col];
  const float bE2a = biasS[1 * 32 + col], bE2b = biasS[1 * 32 + 16 + col];
  const float bE3  = biasS[2 * 32 + col];
  const float bN1a = biasS[3 * 32 + col], bN1b = biasS[3 * 32 + 16 + col];
  const float bN2  = biasS[4 * 32 + col];
  const float bR1a = biasS[5 * 32 + col], bR1b = biasS[5 * 32 + 16 + col];
  const float bR2  = biasS[6 * 32 + col];
  float Mrow[8][8];
#pragma unroll
  for (int n = 0; n < 8; ++n)
#pragma unroll
    for (int m = 0; m < 8; ++m) Mrow[n][m] = MtotS[n * 8 + m];
  lds_fence();

  const int nTiles = (Bn + 15) >> 4;
  const int gwave = blockIdx.x * WPB + wave;
  const int wstride = gridDim.x * WPB;

  for (int tile = gwave; tile < nTiles; tile += wstride) {
    const int base = tile << 4;

    // ---------------- encoder (16 batches = M rows) ----------------
    v16h a0;
#pragma unroll
    for (int i = 0; i < 16; ++i) a0[i] = (_Float16)0.f;
    if (lane < 16) {
      int bb = base + lane; if (bb >= Bn) bb = Bn - 1;
      const float4 xv = *(const float4*)(x + (long)bb * 60 + 56);  // x[bb, L-1, :]
      a0[0] = (_Float16)xv.x; a0[1] = (_Float16)xv.y;
      a0[2] = (_Float16)xv.z; a0[3] = (_Float16)xv.w;
    }
    v8f c0 = wmma16(a0, gatherB(wt + 0 * 512), splat8(bE1a));
    v8f c1 = wmma16(a0, gatherB(wt + 1 * 512), splat8(bE1b));
    storePairTanh(mytB, c0, c1);
    lds_fence();

    v16h a1 = gatherA(mytB, 32);
    c0 = wmma16(a1, gatherB(wt + 2 * 512), splat8(bE2a));
    c1 = wmma16(a1, gatherB(wt + 3 * 512), splat8(bE2b));
    storePairTanh(mytB, c0, c1);
    lds_fence();

    v16h a2 = gatherA(mytB, 32);
    v8f cz = wmma16(a2, gatherB(wt + 4 * 512), splat8(bE3));
    if (col < 3) {
#pragma unroll
      for (int v = 0; v < 8; ++v) myzb[(v + 8 * hi) * 4 + col] = cz[v];
    }
    lds_fence();

    // --------- bilinear scatter + fused node mixing (branch-free VALU) ---------
    const int m = col;  // lanes 16..31 duplicate lanes 0..15, harmless
    const float z0 = myzb[m * 4 + 0], z1 = myzb[m * 4 + 1], z2 = myzb[m * 4 + 2];
    const float rr = fast_sigmoid(z0);                 // * (R-1), R=2
    const float pp = (z1 + PI_F) / (2.f * PI_F) * 4.f;
    const float rf = truncf(rr), pf = truncf(pp);
    const float dr = rr - rf, dp = pp - pf;
    int nodeC[4]; float wC[4];
#pragma unroll
    for (int c = 0; c < 4; ++c) {
      const int dro = c >> 1, dpo = c & 1;
      const float wr = dro ? dr : 1.f - dr;
      const float wp = dpo ? dp : 1.f - dp;
      const float w = wr * wp;
      int ri = (int)rf + dro; if (ri > 1) ri = 1;
      int pi_ = ((int)pf + dpo) % 4; if (pi_ < 0) pi_ += 4;
      nodeC[c] = (w > 0.f) ? (ri * 4 + pi_) : -1;  // -1 = masked out
      wC[c] = w;
    }
    float hl[8][4];
#pragma unroll
    for (int n = 0; n < 8; ++n) {
      float cnt = 0.f, wsum = 0.f;
#pragma unroll
      for (int c = 0; c < 4; ++c) {
        const bool hit = (nodeC[c] == n);
        cnt  += hit ? 1.f   : 0.f;
        wsum += hit ? wC[c] : 0.f;
      }
      hl[n][0] = cnt * z0; hl[n][1] = cnt * z1; hl[n][2] = cnt * z2; hl[n][3] = wsum;
    }
    float h2[8][4];
#pragma unroll
    for (int n = 0; n < 8; ++n)
#pragma unroll
      for (int e = 0; e < 4; ++e) {
        float s = 0.f;
#pragma unroll
        for (int mm = 0; mm < 8; ++mm) s += Mrow[n][mm] * hl[mm][e];
        h2[n][e] = s;
      }

    // ---------------- node net, one node (= one M-tile) at a time ----------------
    for (int n = 0; n < 8; ++n) {
      if (lane < 16) {
        v4h hv;
        hv[0] = (_Float16)h2[n][0]; hv[1] = (_Float16)h2[n][1];
        hv[2] = (_Float16)h2[n][2]; hv[3] = (_Float16)h2[n][3];
        *(v4h*)(mytA + m * 32) = hv;   // single ds_store_b64
      }
      lds_fence();
      v16h an = gatherA(mytA, 32);
      v8f n0 = wmma16(an, wN1a, splat8(bN1a));
      v8f n1 = wmma16(an, wN1b, splat8(bN1b));
      storePairTanh(mytB, n0, n1);
      lds_fence();
      v16h an2 = gatherA(mytB, 32);
      v8f n2 = wmma16(an2, wN2, splat8(bN2));
      if (col < 12) {
#pragma unroll
        for (int v = 0; v < 8; ++v)
          myh3[(v + 8 * hi) * 96 + n * 12 + col] = (_Float16)n2[v];
      }
      lds_fence();
    }

    // ---------------- readout: 16x96 @ 96x24, then 24->4 ----------------
    v8f r0 = splat8(bR1a), r1 = splat8(bR1b);
#pragma unroll
    for (int kt = 0; kt < 3; ++kt) {
      v16h ar = gatherA(myh3 + kt * 32, 96);
      r0 = wmma16(ar, gatherB(wt + (8 + kt * 2) * 512), r0);
      r1 = wmma16(ar, gatherB(wt + (9 + kt * 2) * 512), r1);
    }
    storePairTanh(mytB, r0, r1);
    lds_fence();
    v16h ao = gatherA(mytB, 32);
    v8f oacc = wmma16(ao, gatherB(wt + 14 * 512), splat8(bR2));
    if (col < 4) {
#pragma unroll
      for (int v = 0; v < 8; ++v) {
        const int bb = base + v + 8 * hi;
        if (bb < Bn) out[(long)bb * 4 + col] = oacc[v];
      }
    }
  }
}

extern "C" void kernel_launch(void* const* d_in, const int* in_sizes, int n_in,
                              void* d_out, int out_size, void* d_ws, size_t ws_size,
                              hipStream_t stream) {
  const float* x        = (const float*)d_in[0];
  const float* enc_w1   = (const float*)d_in[1];
  const float* enc_b1   = (const float*)d_in[2];
  const float* enc_w2   = (const float*)d_in[3];
  const float* enc_b2   = (const float*)d_in[4];
  const float* enc_w3   = (const float*)d_in[5];
  const float* enc_b3   = (const float*)d_in[6];
  const float* node_w1  = (const float*)d_in[7];
  const float* node_b1  = (const float*)d_in[8];
  const float* node_w2  = (const float*)d_in[9];
  const float* node_b2  = (const float*)d_in[10];
  const float* ang_l    = (const float*)d_in[11];
  const float* rad_l    = (const float*)d_in[12];
  const float* read_w1  = (const float*)d_in[13];
  const float* read_b1  = (const float*)d_in[14];
  const float* read_w2  = (const float*)d_in[15];
  const float* read_b2  = (const float*)d_in[16];
  float* out = (float*)d_out;

  const int B = in_sizes[0] / 60;  // x is [B, 15, 4]
  const int nTiles = (B + 15) / 16;
  int blocks = (nTiles + WPB - 1) / WPB;
  if (blocks < 1) blocks = 1;
  if (blocks > 4096) blocks = 4096;

  topo_fwd<<<blocks, BLOCK, 0, stream>>>(
      x, enc_w1, enc_b1, enc_w2, enc_b2, enc_w3, enc_b3,
      node_w1, node_b1, node_w2, node_b2, ang_l, rad_l,
      read_w1, read_b1, read_w2, read_b2, out, B);
}
